// GCNT_17721035063382
// MI455X (gfx1250) — matrix operand order repaired and checked by reading, not compiled, not run on hardware
//
#include <hip/hip_runtime.h>

// ---------------- types ----------------
typedef __bf16 bf16_t;
typedef __attribute__((ext_vector_type(16))) __bf16 v16bf;
typedef __attribute__((ext_vector_type(8)))  __bf16 v8bf;
typedef __attribute__((ext_vector_type(4)))  __bf16 v4bf;
typedef __attribute__((ext_vector_type(8)))  float   v8f;
typedef __attribute__((ext_vector_type(4)))  float   v4f;
typedef __attribute__((ext_vector_type(4)))  int     v4i;

#define LPITCH 40   // bf16 elems per LDS row: 32 data + 8 pad (80B, kills b128 bank conflicts)
#define FPITCH 36   // f32  elems per LDS row: 32 data + 4 pad (144B, distinct banks for 16 lanes)

// ---- CDNA5 async global->LDS path (probe-guarded; falls back to reg pipeline) ----
#if __has_builtin(__builtin_amdgcn_global_load_async_to_lds_b128)
#define HAVE_ASYNC 1
typedef __attribute__((address_space(1))) v4i gv4i;   // global int4*
typedef __attribute__((address_space(3))) v4i lv4i;   // LDS int4*
__device__ __forceinline__ void async_b128(const void* g, void* l) {
  __builtin_amdgcn_global_load_async_to_lds_b128((gv4i*)g, (lv4i*)l, 0, 0);
}
#else
#define HAVE_ASYNC 0
#endif

template<int N>
__device__ __forceinline__ void wait_async() {
#if __has_builtin(__builtin_amdgcn_s_wait_asynccnt)
  __builtin_amdgcn_s_wait_asynccnt(N);
#else
  asm volatile("s_wait_asynccnt %0" :: "i"(N) : "memory");
#endif
}

__device__ __forceinline__ v16bf ld_frag(const bf16_t* rowbase, int off0, int off1) {
  v8bf lo = *(const v8bf*)((const char*)rowbase + off0);
  v8bf hi = *(const v8bf*)((const char*)rowbase + off1);
  return __builtin_shufflevector(lo, hi, 0,1,2,3,4,5,6,7,8,9,10,11,12,13,14,15);
}

__device__ __forceinline__ v8f wmma_bf16(v16bf a, v16bf b, v8f c) {
  // 8 args: (neg_a, A, neg_b, B, c_mod, C, reuse_a, reuse_b)
  return __builtin_amdgcn_wmma_f32_16x16x32_bf16(false, a, false, b, (short)0, c, false, false);
}

// =====================================================================
// Small GEMM (steps 1 & 3):  C = A(f32|bf16 [M,K]) @ B(f32 [K,Nc])
// Output written TRANSPOSED as bf16 with row pitch Mld (>= M): CT[Nc][Mld].
// Columns m in [M, Mld) are written as ZEROS (k-pad for the adj GEMM).
// BM=BN=64, BK=32, 128 threads = 4 waves, each wave a 32x32 subtile.
// =====================================================================
template<bool A_BF16>
__global__ __launch_bounds__(128)
void gemm_small_kernel(const void* __restrict__ Aptr, const float* __restrict__ B,
                       bf16_t* __restrict__ CT, int M, int K, int Nc, int Mld)
{
  __shared__ __align__(16) bf16_t lsA[64 * LPITCH];
  __shared__ __align__(16) bf16_t lsB[64 * LPITCH];
  __shared__ __align__(16) float  lsC[64 * 68];

  const int tid  = threadIdx.x;
  const int lane = tid & 31;
  const int wid  = tid >> 5;
  const int wm   = wid & 1;
  const int wn   = wid >> 1;
  const int m0   = blockIdx.y * 64;
  const int n0   = blockIdx.x * 64;

  v8f acc[2][2] = {};

  const int nsteps = K >> 5;   // K is 512 or 128: exact
  for (int s = 0; s < nsteps; ++s) {
    const int k0 = s << 5;
    // ---- stage A tile (64 rows x 32 k), row-major in LDS; OOB rows -> zeros ----
    {
      const int ar = tid >> 1;
      const int c0 = (tid & 1) * 16;
      const int gm = m0 + ar;
      bf16_t* d = &lsA[ar * LPITCH + c0];
      if (A_BF16) {
        const bf16_t* Ab = (const bf16_t*)Aptr;
        uint4 u0 = make_uint4(0,0,0,0), u1 = make_uint4(0,0,0,0);
        if (gm < M) {
          const bf16_t* p = Ab + (size_t)gm * K + k0 + c0;
          u0 = *(const uint4*)p;
          u1 = *(const uint4*)(p + 8);
        }
        *(uint4*)d = u0;
        *(uint4*)(d + 8) = u1;
      } else {
        const float* Af = (const float*)Aptr;
        const v4f vz = {0.f,0.f,0.f,0.f};
        v4f f[4] = {vz, vz, vz, vz};
        if (gm < M) {
          const float* p = Af + (size_t)gm * K + k0 + c0;
          #pragma unroll
          for (int i = 0; i < 4; ++i) f[i] = *(const v4f*)(p + 4*i);
        }
        v8bf p0, p1;
        #pragma unroll
        for (int i = 0; i < 4; ++i) {
          p0[i]   = (bf16_t)f[0][i]; p0[4+i] = (bf16_t)f[1][i];
          p1[i]   = (bf16_t)f[2][i]; p1[4+i] = (bf16_t)f[3][i];
        }
        *(v8bf*)d = p0;
        *(v8bf*)(d + 8) = p1;
      }
    }
    // ---- stage B tile (32 k x 64 n): coalesced row loads, transpose into lsB[n][k] ----
    {
      const int kr = tid >> 2;          // 0..31
      const int cn = (tid & 3) * 16;    // 0,16,32,48
      const float* p = B + (size_t)(k0 + kr) * Nc + n0 + cn;
      #pragma unroll
      for (int i = 0; i < 4; ++i) {
        v4f f = *(const v4f*)(p + 4*i);
        #pragma unroll
        for (int q = 0; q < 4; ++q)
          lsB[(cn + 4*i + q) * LPITCH + kr] = (bf16_t)f[q];
      }
    }
    __syncthreads();
    // ---- compute: 4 WMMA per wave per K-step ----
    {
      const int la    = lane & 15;
      const int offA0 = (lane < 16) ? 0 : 16;   // bytes: K0-7 / K8-15
      const int offB0 = (lane < 16) ? 0 : 16;   // bytes: K0-7 / K8-15
      v16bf bfr[2];
      #pragma unroll
      for (int fn = 0; fn < 2; ++fn)
        bfr[fn] = ld_frag(&lsB[(wn*32 + fn*16 + la) * LPITCH], offB0, offB0 + 32);
      #pragma unroll
      for (int fm = 0; fm < 2; ++fm) {
        v16bf afr = ld_frag(&lsA[(wm*32 + fm*16 + la) * LPITCH], offA0, offA0 + 32);
        #pragma unroll
        for (int fn = 0; fn < 2; ++fn)
          acc[fm][fn] = wmma_bf16(afr, bfr[fn], acc[fm][fn]);
      }
    }
    __syncthreads();
  }

  // ---- LDS transpose epilogue: coalesced bf16 stores of CT[n][m] ----
  {
    const int la = lane & 15;
    const int rb = (lane >> 4) << 3;    // lanes 16-31 hold rows +8 (ISA C/D layout)
    #pragma unroll
    for (int fm = 0; fm < 2; ++fm)
      #pragma unroll
      for (int fn = 0; fn < 2; ++fn) {
        const int c = wn*32 + fn*16 + la;
        #pragma unroll
        for (int j = 0; j < 8; ++j)
          lsC[(wm*32 + fm*16 + rb + j) * 68 + c] = acc[fm][fn][j];
      }
  }
  __syncthreads();
  {
    const int nn   = tid >> 1;
    const int mseg = (tid & 1) * 32;
    const size_t gbase = (size_t)(n0 + nn) * Mld + m0 + mseg;
    #pragma unroll
    for (int c = 0; c < 4; ++c) {
      v8bf pk;
      #pragma unroll
      for (int i = 0; i < 8; ++i)
        pk[i] = (bf16_t)lsC[(mseg + c*8 + i) * 68 + nn];
      // rows m >= M carry zero accumulators -> zero pad lands in [M, Mld)
      if (m0 + mseg + c*8 + 8 <= Mld)   // Mld % 8 == 0
        *(v8bf*)(CT + gbase + c*8) = pk;
    }
  }
}

// =====================================================================
// adj GEMM (steps 2 & 4):  C = adj(f32 [M,K=M]) @ Bt(bf16 [Ncols][Kpad])^T
// BM=BN=128, BK=32, 256 threads = 8 waves (4 along M x 2 along N),
// each wave 32x64 => 8 wmma accumulators.
// HAVE_ASYNC: tiles DMA'd straight to LDS (global_load_async_to_lds_b128,
//   ASYNCcnt), adj kept f32 in LDS and converted during fragment build.
// else: register-staged double-buffered pipeline.
// Bt's k-pad columns [K, Kpad) are zero, so clamped/garbage adj columns
// in the k-tail contribute exactly zero.
// =====================================================================
template<bool FUSE>
__global__ __launch_bounds__(256)
void adj_gemm_kernel(const float* __restrict__ A, const bf16_t* __restrict__ Bt,
                     float* __restrict__ Cout, const float* __restrict__ bias,
                     int M, int Ncols, int K, int Kpad)
{
  const int tid  = threadIdx.x;
  const int lane = tid & 31;
  const int wid  = tid >> 5;
  const int wm   = wid & 3;     // 4 waves along M (32 rows each)
  const int wn   = wid >> 2;    // 2 waves along N (64 cols each)
  const int m0   = blockIdx.y * 128;
  const int n0   = blockIdx.x * 128;

  const int nsteps = (K + 31) >> 5;
  const int nz     = (int)gridDim.z;
  const int chunk  = (nsteps + nz - 1) / nz;
  const int s0     = (int)blockIdx.z * chunk;
  const int s1     = (s0 + chunk < nsteps) ? (s0 + chunk) : nsteps;

  v8f acc[2][4] = {};

  const int l_r  = tid >> 1;          // row within tile: 0..127 (A rows / Bt rows)
  const int l_c0 = (tid & 1) * 16;    // k sub-segment: 0 or 16
  const int la   = lane & 15;

#if HAVE_ASYNC
  {
    __shared__ __align__(16) float  lsAf[2][128 * FPITCH];
    __shared__ __align__(16) bf16_t lsB [2][128 * LPITCH];

    auto issue = [&](int s, int buf) {
      const int k0 = s << 5;
      const int kc = k0 + l_c0;
      {   // adj tile: raw f32, 4 x async b128; clamp OOB addresses (k-tail is
          // neutralized by Bt's zero pad, OOB rows are never stored)
        const int gm  = (m0 + l_r < M) ? (m0 + l_r) : (M - 1);
        const int kcc = (kc + 16 <= K) ? kc : 0;
        const float* gp = A + (size_t)gm * K + kcc;
        float* lp = &lsAf[buf][l_r * FPITCH + l_c0];
        #pragma unroll
        for (int i = 0; i < 4; ++i)
          async_b128(gp + 4*i, lp + 4*i);
      }
      {   // Bt tile: bf16, rows of length Kpad (zero-padded), 2 x async b128
        const bf16_t* gp = Bt + (size_t)(n0 + l_r) * Kpad + kc;
        bf16_t* lp = &lsB[buf][l_r * LPITCH + l_c0];
        async_b128(gp,     lp);
        async_b128(gp + 8, lp + 8);
      }
    };

    if (s0 < s1) issue(s0, 0);
    wait_async<0>();
    __syncthreads();

    for (int s = s0; s < s1; ++s) {
      const int buf = (s - s0) & 1;
      const bool pre = (s + 1 < s1);
      if (pre) issue(s + 1, buf ^ 1);
      {
        const float*  lA = lsAf[buf];
        const bf16_t* lB = lsB[buf];
        const int fo0   = (lane < 16) ? 0  : 32;   // bytes: f32 K0-7 / K8-15
        const int fo1   = (lane < 16) ? 64 : 96;   // bytes: f32 K16-23 / K24-31
        const int offB0 = (lane < 16) ? 0  : 16;
        v16bf bfr[4];
        #pragma unroll
        for (int fn = 0; fn < 4; ++fn)
          bfr[fn] = ld_frag(&lB[(wn*64 + fn*16 + la) * LPITCH], offB0, offB0 + 32);
        #pragma unroll
        for (int fm = 0; fm < 2; ++fm) {
          const char* rp = (const char*)&lA[(wm*32 + fm*16 + la) * FPITCH];
          v4f a0 = *(const v4f*)(rp + fo0);
          v4f a1 = *(const v4f*)(rp + fo0 + 16);
          v4f a2 = *(const v4f*)(rp + fo1);
          v4f a3 = *(const v4f*)(rp + fo1 + 16);
          v16bf afr;
          #pragma unroll
          for (int i = 0; i < 4; ++i) {
            afr[i]    = (bf16_t)a0[i];
            afr[4+i]  = (bf16_t)a1[i];
            afr[8+i]  = (bf16_t)a2[i];
            afr[12+i] = (bf16_t)a3[i];
          }
          #pragma unroll
          for (int fn = 0; fn < 4; ++fn)
            acc[fm][fn] = wmma_bf16(afr, bfr[fn], acc[fm][fn]);
        }
      }
      if (pre) wait_async<6>();   // only the 6 just-issued may remain in flight
      else     wait_async<0>();
      __syncthreads();            // makes other waves' async LDS writes visible
    }
  }
#else
  {
    __shared__ __align__(16) bf16_t lsA[2][128 * LPITCH];
    __shared__ __align__(16) bf16_t lsB[2][128 * LPITCH];

    const v4f vz = {0.f,0.f,0.f,0.f};
    v4f  ra[4] = {vz, vz, vz, vz};
    uint4 rb0 = make_uint4(0,0,0,0), rb1 = make_uint4(0,0,0,0);

    auto g_load = [&](int s) {
      const int k0 = s << 5;
      const int kc = k0 + l_c0;
      {   // adj tile: f32, row-major, lda == K == M, zero-fill OOB
        const int gm = m0 + l_r;
        const float* p = A + (size_t)gm * K + kc;
        #pragma unroll
        for (int i = 0; i < 4; ++i) {
          const bool ok = (gm < M) && (kc + 4*i + 4 <= K);
          ra[i] = ok ? *(const v4f*)(p + 4*i) : vz;
        }
      }
      {   // Bt tile: zero-padded rows of length Kpad -> unconditional loads
        const bf16_t* p = Bt + (size_t)(n0 + l_r) * Kpad + kc;
        rb0 = *(const uint4*)(p);
        rb1 = *(const uint4*)(p + 8);
      }
    };

    auto l_store = [&](int buf) {
      {
        bf16_t* d = &lsA[buf][l_r * LPITCH + l_c0];
        v8bf p0, p1;
        #pragma unroll
        for (int i = 0; i < 4; ++i) {
          p0[i]   = (bf16_t)ra[0][i]; p0[4+i] = (bf16_t)ra[1][i];
          p1[i]   = (bf16_t)ra[2][i]; p1[4+i] = (bf16_t)ra[3][i];
        }
        *(v8bf*)d = p0;
        *(v8bf*)(d + 8) = p1;
      }
      {
        bf16_t* d = &lsB[buf][l_r * LPITCH + l_c0];
        *(uint4*)d       = rb0;
        *(uint4*)(d + 8) = rb1;
      }
    };

    if (s0 < s1) { g_load(s0); l_store(0); }
    __syncthreads();

    for (int s = s0; s < s1; ++s) {
      const int buf = (s - s0) & 1;
      if (s + 1 < s1) g_load(s + 1);
      {
        const bf16_t* lA = lsA[buf];
        const bf16_t* lB = lsB[buf];
        const int offA0 = (lane < 16) ? 0 : 16;
        const int offB0 = (lane < 16) ? 0 : 16;
        v16bf bfr[4];
        #pragma unroll
        for (int fn = 0; fn < 4; ++fn)
          bfr[fn] = ld_frag(&lB[(wn*64 + fn*16 + la) * LPITCH], offB0, offB0 + 32);
        #pragma unroll
        for (int fm = 0; fm < 2; ++fm) {
          v16bf afr = ld_frag(&lA[(wm*32 + fm*16 + la) * LPITCH], offA0, offA0 + 32);
          #pragma unroll
          for (int fn = 0; fn < 4; ++fn)
            acc[fm][fn] = wmma_bf16(afr, bfr[fn], acc[fm][fn]);
        }
      }
      if (s + 1 < s1) l_store(buf ^ 1);
      __syncthreads();
    }
  }
#endif

  float* Cz = FUSE ? Cout : (Cout + (size_t)blockIdx.z * (size_t)M * Ncols);
  const int rb = (lane >> 4) << 3;
  #pragma unroll
  for (int fm = 0; fm < 2; ++fm)
    #pragma unroll
    for (int fn = 0; fn < 4; ++fn) {
      const int gn = n0 + wn*64 + fn*16 + la;
      const float bv = FUSE ? bias[gn] : 0.0f;
      #pragma unroll
      for (int j = 0; j < 8; ++j) {
        const int gm = m0 + wm*32 + fm*16 + rb + j;
        if (gm < M)
          Cz[(size_t)gm * Ncols + gn] = acc[fm][fn][j] + bv;
      }
    }
}

// =====================================================================
// Split-K reduction + bias + ReLU -> bf16 H   (NSPLIT = 4 partials)
// =====================================================================
__global__ __launch_bounds__(256)
void epi_relu_bias_kernel(const float* __restrict__ P, const float* __restrict__ bias,
                          bf16_t* __restrict__ H, int M, int NH)
{
  const int idx = (blockIdx.x * 256 + threadIdx.x) * 4;
  const int total = M * NH;
  if (idx >= total) return;
  const size_t stride = (size_t)total;
  v4f s = *(const v4f*)(P + idx);
  s += *(const v4f*)(P + stride     + idx);
  s += *(const v4f*)(P + 2*stride   + idx);
  s += *(const v4f*)(P + 3*stride   + idx);
  const int j = idx & (NH - 1);   // NH = 128 (power of 2), idx % 4 == 0
  v4bf r;
  #pragma unroll
  for (int i = 0; i < 4; ++i) {
    float v = s[i] + bias[j + i];
    r[i] = (bf16_t)fmaxf(v, 0.0f);
  }
  *(v4bf*)(H + idx) = r;
}

// =====================================================================
extern "C" void kernel_launch(void* const* d_in, const int* in_sizes, int n_in,
                              void* d_out, int out_size, void* d_ws, size_t ws_size,
                              hipStream_t stream)
{
  (void)in_sizes; (void)n_in; (void)out_size; (void)ws_size;
  const int M = 10000, NF = 512, NH = 128;
  const int Kpad = ((M + 31) / 32) * 32;   // 10016: k-padded Bt row pitch

  const float* x   = (const float*)d_in[0];
  const float* adj = (const float*)d_in[1];
  const float* W1  = (const float*)d_in[2];
  const float* b1  = (const float*)d_in[3];
  const float* W2  = (const float*)d_in[4];
  const float* b2  = (const float*)d_in[5];
  float* out = (float*)d_out;

  // workspace carve-out (~36 MB), fully rewritten every launch
  char* ws = (char*)d_ws;
  size_t off = 0;
  auto carve = [&](size_t bytes) -> void* {
    void* p = ws + off;
    off = (off + bytes + 255) & ~(size_t)255;
    return p;
  };
  bf16_t* XWT = (bf16_t*)carve((size_t)NH * Kpad * sizeof(bf16_t));   // [128][10016]
  bf16_t* HWT = (bf16_t*)carve((size_t)NF * Kpad * sizeof(bf16_t));   // [512][10016]
  bf16_t* H   = (bf16_t*)carve((size_t)M * NH * sizeof(bf16_t));      // [10000][128]
  float*  P2  = (float*) carve((size_t)4 * M * NH * sizeof(float));   // 4 split-K partials

  const int mt64  = (Kpad + 63) / 64;   // 157 (covers the zero pad too)
  const int mt128 = (M + 127) / 128;    // 79

  // 1. XW^T = (x @ W1)^T, bf16, k-padded with zeros
  gemm_small_kernel<false><<<dim3(NH/64, mt64), 128, 0, stream>>>(x, W1, XWT, M, NF, NH, Kpad);
  // 2. P2[z] = partial adj @ XW  (split-K = 4 for occupancy: 79 -> 316 WGs)
  adj_gemm_kernel<false><<<dim3(NH/128, mt128, 4), 256, 0, stream>>>(adj, XWT, P2, nullptr, M, NH, M, Kpad);
  // 3. H = bf16(relu(sum_z P2 + b1))
  epi_relu_bias_kernel<<<(M*NH/4 + 255)/256, 256, 0, stream>>>(P2, b1, H, M, NH);
  // 4. HW^T = (H @ W2)^T, bf16, k-padded with zeros
  gemm_small_kernel<true><<<dim3(NF/64, mt64), 128, 0, stream>>>(H, W2, HWT, M, NH, NF, Kpad);
  // 5. out = adj @ HW + b2  (316 WGs, bias fused, f32 direct to d_out)
  adj_gemm_kernel<true><<<dim3(NF/128, mt128, 1), 256, 0, stream>>>(adj, HWT, out, b2, M, NF, M, Kpad);
}
